// DateConversionModel_69312182223013
// MI455X (gfx1250) — matrix operand order repaired and checked by reading, not compiled, not run on hardware
//
#include <hip/hip_runtime.h>
#include <hip/hip_bf16.h>
#include <math.h>

typedef __bf16 bf16;
typedef __attribute__((ext_vector_type(16))) __bf16 v16bf;
typedef __attribute__((ext_vector_type(8)))  float  v8f;

// ---- model dims ----
constexpr int Bn  = 32;
constexpr int Sn  = 128;
constexpr int Tn  = 10;
constexpr int Hn  = 768;
constexpr int Ln  = 12;
constexpr int NHn = 12;
constexpr int FFn = 3072;
constexpr int Cn  = 16;
constexpr int DHn = 64;
constexpr int CLSn = 101;
constexpr int NTOK = Bn * Sn;     // 4096 rows

__device__ __forceinline__ v8f wmma_bf16(v16bf a, v16bf b, v8f c) {
  // D(f32 16x16) = A(16x32 bf16) * B(32x16 bf16) + C
  return __builtin_amdgcn_wmma_f32_16x16x32_bf16(false, a, false, b, (short)0, c,
                                                 false, false);
}

// A-fragment K index for element e, lane-half `half` (ISA 05_wmma §7.12.2,
// 16-bit A 16x32: VGPR j holds K = {2j, 2j+1} (+8 for j>=4), +8 for lanes 16..31)
__device__ __forceinline__ int a_kidx(int e, int half) {
  int j = e >> 1, r = e & 1;
  return 2 * j + ((j >= 4) ? 8 : 0) + 8 * half + r;
}

// ---- CDNA5 async global->LDS copy (ASYNCcnt), ISA 08_async_tensor §4 ----
// per-lane: 16 bytes MEM[gaddr] -> LDS[lds_byte_off]; both must be 16B aligned.
__device__ __forceinline__ void async_cp16(unsigned lds_off, const void* g) {
  asm volatile("global_load_async_to_lds_b128 %0, %1, off"
               :: "v"(lds_off), "v"((unsigned long long)(uintptr_t)g)
               : "memory");
}
__device__ __forceinline__ void wait_async0() {
  asm volatile("s_wait_asynccnt 0x0" ::: "memory");
}
__device__ __forceinline__ unsigned lds_off32(const void* p) {
  // generic pointer to LDS: low 32 bits are the LDS byte offset (flat aperture)
  return (unsigned)(uintptr_t)p;
}

// ---------------- f32 -> bf16 conversion ----------------
__global__ __launch_bounds__(256) void cvt_bf16_k(const float* __restrict__ s,
                                                  bf16* __restrict__ d, int n) {
  int i = blockIdx.x * 256 + threadIdx.x;
  if (i < n) d[i] = (bf16)s[i];
}

// ---------------- WMMA GEMM: C[MxN] = A[MxK](bf16) * B[KxN](bf16) + bias ----
// 256 threads (8 wave32), block tile 128x128, wave tile 64x32, K-step 32,
// async-double-buffered LDS staging. epi: 0 = none, 1 = exact GELU
constexpr int AP = 40;    // A LDS pitch (halfwords): 80B rows, 16B aligned
constexpr int BP = 136;   // B LDS pitch (halfwords): 272B rows, 16B aligned

__global__ __launch_bounds__(256)
void gemm_bf16_k(const bf16* __restrict__ A, const bf16* __restrict__ Bw,
                 const float* __restrict__ bias,
                 float* __restrict__ Cf, bf16* __restrict__ Cb,
                 int M, int N, int K, int epi) {
  __shared__ bf16 As[2][128][AP];
  __shared__ bf16 Bs[2][32][BP];

  const int tid  = threadIdx.x;
  const int lane = tid & 31;
  const int wid  = tid >> 5;
  const int lm   = lane & 15;
  const int lh   = lane >> 4;
  const int bm   = blockIdx.y * 128;
  const int bn   = blockIdx.x * 128;
  const int wm   = (wid >> 2) * 64;   // 0 / 64
  const int wn   = (wid & 3) * 32;    // 0 / 32 / 64 / 96

  v8f acc[4][2];
#pragma unroll
  for (int mt = 0; mt < 4; ++mt)
#pragma unroll
    for (int nt = 0; nt < 2; ++nt)
#pragma unroll
      for (int r = 0; r < 8; ++r) acc[mt][nt][r] = 0.f;

  // async stage of one K-step tile pair into buffer `buf`
  auto stage = [&](int k0, int buf) {
    // A tile 128x32: 512 x 16B chunks (2 per thread)
#pragma unroll
    for (int it = 0; it < 2; ++it) {
      int i = tid + it * 256;
      int r = i >> 2, c = (i & 3) * 8;
      async_cp16(lds_off32(&As[buf][r][c]), A + (size_t)(bm + r) * K + k0 + c);
    }
    // B tile 32x128: 512 x 16B chunks (2 per thread)
#pragma unroll
    for (int it = 0; it < 2; ++it) {
      int i = tid + it * 256;
      int r = i >> 4, c = (i & 15) * 8;
      async_cp16(lds_off32(&Bs[buf][r][c]), Bw + (size_t)(k0 + r) * N + bn + c);
    }
  };

  stage(0, 0);
  wait_async0();
  __syncthreads();

  int cur = 0;
  for (int k0 = 0; k0 < K; k0 += 32) {
    if (k0 + 32 < K) stage(k0 + 32, cur ^ 1);   // prefetch next tile (async)

    v16bf bfrag[2];
#pragma unroll
    for (int nt = 0; nt < 2; ++nt)
#pragma unroll
      for (int e = 0; e < 16; ++e)
        bfrag[nt][e] = Bs[cur][a_kidx(e, lh)][wn + nt * 16 + lm];  // B: lane = N, elems = K

#pragma unroll
    for (int mt = 0; mt < 4; ++mt) {
      v16bf afrag;
      int m = wm + mt * 16 + lm;
#pragma unroll
      for (int e = 0; e < 16; ++e) afrag[e] = As[cur][m][a_kidx(e, lh)];
      acc[mt][0] = wmma_bf16(afrag, bfrag[0], acc[mt][0]);
      acc[mt][1] = wmma_bf16(afrag, bfrag[1], acc[mt][1]);
    }

    wait_async0();      // next buffer fully landed in LDS
    __syncthreads();    // everyone done reading `cur`, `nxt` visible to all
    cur ^= 1;
  }

  // epilogue: C/D layout — lane = N, VGPR r -> M = r + 8*half
#pragma unroll
  for (int mt = 0; mt < 4; ++mt)
#pragma unroll
    for (int nt = 0; nt < 2; ++nt) {
      int n = bn + wn + nt * 16 + lm;
      float bv = bias[n];
#pragma unroll
      for (int r = 0; r < 8; ++r) {
        int m = bm + wm + mt * 16 + r + 8 * lh;
        float v = acc[mt][nt][r] + bv;
        if (epi == 1) v = 0.5f * v * (1.f + erff(v * 0.70710678118654752f));
        size_t o = (size_t)m * N + n;
        if (Cf) Cf[o] = v;
        if (Cb) Cb[o] = (bf16)v;
      }
    }
}

// ---------------- fused attention per (b, h) ----------------
// 128 threads (4 wave32); dynamic LDS: K,V bf16 (pitch 72 = 144B, 16B aligned),
// S f32 (pitch 129), P bf16 (pitch 136)
constexpr int KVP = 72, SP = 129, PP = 136;
constexpr size_t ATT_LDS =
    (size_t)128 * KVP * 2 * 2 + (size_t)128 * SP * 4 + (size_t)128 * PP * 2;

__global__ __launch_bounds__(128)
void attention_k(const bf16* __restrict__ q, const bf16* __restrict__ k,
                 const bf16* __restrict__ v, const int* __restrict__ mask,
                 bf16* __restrict__ ctx) {
  extern __shared__ char smem[];
  bf16*  Kl = (bf16*)smem;
  bf16*  Vl = Kl + 128 * KVP;
  float* Sl = (float*)(Vl + 128 * KVP);
  bf16*  Pl = (bf16*)(Sl + 128 * SP);

  const int bh = blockIdx.x;
  const int b  = bh / NHn, h = bh % NHn;
  const int tid = threadIdx.x, lane = tid & 31, w = tid >> 5;
  const int lm = lane & 15, lh = lane >> 4;
  const size_t base = ((size_t)b * Sn) * Hn + (size_t)h * DHn;

  // async-stage K and V (128x64 bf16 each = 8 x 16B chunks per row)
  for (int i = tid; i < 2048; i += 128) {
    int which = i & 1;           // interleave K / V issues
    int idx = i >> 1;            // 0..1023
    int s = idx >> 3, c = (idx & 7) * 8;
    const bf16* g = (which ? v : k) + base + (size_t)s * Hn + c;
    bf16* l = (which ? Vl : Kl) + s * KVP + c;
    async_cp16(lds_off32(l), g);
  }
  wait_async0();
  __syncthreads();

  const float scale = 0.125f;   // 1/sqrt(64)

  // phase 1: S = scale * Q K^T + mask   (wave w handles query rows w*32..w*32+31)
#pragma unroll
  for (int mt = 0; mt < 2; ++mt) {
    int m_off = w * 32 + mt * 16;
    v16bf aq[2];
#pragma unroll
    for (int ks = 0; ks < 2; ++ks)
#pragma unroll
      for (int e = 0; e < 16; ++e)
        aq[ks][e] = q[base + (size_t)(m_off + lm) * Hn + ks * 32 + a_kidx(e, lh)];
#pragma unroll
    for (int nt = 0; nt < 8; ++nt) {
      v8f acc;
#pragma unroll
      for (int r = 0; r < 8; ++r) acc[r] = 0.f;
#pragma unroll
      for (int ks = 0; ks < 2; ++ks) {
        v16bf bk;  // B = K^T : B[kk][n] = K[n][kk]; lane = n, elem = kk
#pragma unroll
        for (int e = 0; e < 16; ++e)
          bk[e] = Kl[(nt * 16 + lm) * KVP + ks * 32 + a_kidx(e, lh)];
        acc = wmma_bf16(aq[ks], bk, acc);
      }
      int n = nt * 16 + lm;
      float mterm = (1.f - (float)mask[b * Sn + n]) * -1.0e9f;
#pragma unroll
      for (int r = 0; r < 8; ++r)
        Sl[(m_off + r + 8 * lh) * SP + n] = acc[r] * scale + mterm;
    }
  }
  __syncthreads();

  // phase 2: softmax, one row per thread
  {
    float* row = Sl + tid * SP;
    float mx = -1e30f;
    for (int j = 0; j < 128; ++j) mx = fmaxf(mx, row[j]);
    float sum = 0.f;
    for (int j = 0; j < 128; ++j) { float e = expf(row[j] - mx); row[j] = e; sum += e; }
    float inv = 1.f / sum;
    bf16* prow = Pl + tid * PP;
    for (int j = 0; j < 128; ++j) prow[j] = (bf16)(row[j] * inv);
  }
  __syncthreads();

  // phase 3: ctx = P V
#pragma unroll
  for (int mt = 0; mt < 2; ++mt) {
    int m_off = w * 32 + mt * 16;
    v16bf ap[4];
#pragma unroll
    for (int ks = 0; ks < 4; ++ks)
#pragma unroll
      for (int e = 0; e < 16; ++e)
        ap[ks][e] = Pl[(m_off + lm) * PP + ks * 32 + a_kidx(e, lh)];
#pragma unroll
    for (int nt = 0; nt < 4; ++nt) {
      v8f acc;
#pragma unroll
      for (int r = 0; r < 8; ++r) acc[r] = 0.f;
#pragma unroll
      for (int ks = 0; ks < 4; ++ks) {
        v16bf bv;
#pragma unroll
        for (int e = 0; e < 16; ++e)
          bv[e] = Vl[(ks * 32 + a_kidx(e, lh)) * KVP + nt * 16 + lm];
        acc = wmma_bf16(ap[ks], bv, acc);
      }
#pragma unroll
      for (int r = 0; r < 8; ++r) {
        int m = m_off + r + 8 * lh;
        int n = nt * 16 + lm;
        ctx[base + (size_t)m * Hn + n] = (bf16)acc[r];
      }
    }
  }
}

// ---------------- LayerNorm helpers (block = 256 threads, one 768-row) -------
__device__ __forceinline__ void ln_write(float* buf, float* red, float partial,
                                         const float* __restrict__ g,
                                         const float* __restrict__ be,
                                         float* __restrict__ y,
                                         bf16* __restrict__ yb) {
  int tid = threadIdx.x;
  red[tid] = partial; __syncthreads();
  for (int o = 128; o > 0; o >>= 1) { if (tid < o) red[tid] += red[tid + o]; __syncthreads(); }
  float mean = red[0] * (1.f / Hn); __syncthreads();
  float vv = 0.f;
  for (int i = tid; i < Hn; i += 256) { float d = buf[i] - mean; vv += d * d; }
  red[tid] = vv; __syncthreads();
  for (int o = 128; o > 0; o >>= 1) { if (tid < o) red[tid] += red[tid + o]; __syncthreads(); }
  float inv = rsqrtf(red[0] * (1.f / Hn) + 1e-12f);
  for (int i = tid; i < Hn; i += 256) {
    float o = (buf[i] - mean) * inv * g[i] + be[i];
    if (y)  y[i]  = o;
    if (yb) yb[i] = (bf16)o;
  }
}

__global__ __launch_bounds__(256)
void embed_ln_k(const int* __restrict__ ids, const float* __restrict__ ew,
                const float* __restrict__ ep, const float* __restrict__ et,
                const float* __restrict__ g, const float* __restrict__ be,
                float* __restrict__ y, bf16* __restrict__ yb) {
  __shared__ float buf[Hn];
  __shared__ float red[256];
  int row = blockIdx.x, s = row % Sn, tid = threadIdx.x;
  int id = ids[row];
  float acc = 0.f;
  for (int i = tid; i < Hn; i += 256) {
    float t = ew[(size_t)id * Hn + i] + ep[(size_t)s * Hn + i] + et[i];
    buf[i] = t; acc += t;
  }
  ln_write(buf, red, acc, g, be, y + (size_t)row * Hn, yb + (size_t)row * Hn);
}

__global__ __launch_bounds__(256)
void add_ln_k(const float* __restrict__ x, const float* __restrict__ p,
              const float* __restrict__ g, const float* __restrict__ be,
              float* __restrict__ y, bf16* __restrict__ yb) {
  __shared__ float buf[Hn];
  __shared__ float red[256];
  int row = blockIdx.x, tid = threadIdx.x;
  const float* xr = x + (size_t)row * Hn;
  const float* pr = p + (size_t)row * Hn;
  float acc = 0.f;
  for (int i = tid; i < Hn; i += 256) { float t = xr[i] + pr[i]; buf[i] = t; acc += t; }
  ln_write(buf, red, acc, g, be, y + (size_t)row * Hn, yb + (size_t)row * Hn);
}

__global__ __launch_bounds__(256)
void decin_ln_k(const float* __restrict__ ew, const float* __restrict__ ep,
                const float* __restrict__ et, const float* __restrict__ g,
                const float* __restrict__ be, float* __restrict__ y) {
  __shared__ float buf[Hn];
  __shared__ float red[256];
  int tid = threadIdx.x;
  float acc = 0.f;
  for (int i = tid; i < Hn; i += 256) {
    float t = ew[(size_t)CLSn * Hn + i] + ep[i] + et[i];
    buf[i] = t; acc += t;
  }
  ln_write(buf, red, acc, g, be, y, nullptr);
}

// ---------------- GRU decoder (tiny; plain VALU) ----------------
__global__ __launch_bounds__(256)
void gather_hidden_k(const float* __restrict__ x, float* __restrict__ h) {
  int idx = blockIdx.x * 256 + threadIdx.x;
  if (idx >= Bn * Hn) return;
  int b = idx / Hn, j = idx % Hn;
  h[idx] = x[((size_t)b * Sn + (Sn - 1)) * Hn + j];
}

__global__ __launch_bounds__(256)
void gru_gemm_k(const float* __restrict__ X, int xstride,
                const float* __restrict__ W, const float* __restrict__ bias,
                float* __restrict__ out) {
  int idx = blockIdx.x * 256 + threadIdx.x;
  if (idx >= Bn * 3 * Hn) return;
  int b = idx / (3 * Hn), j = idx % (3 * Hn);
  const float* xr = X + (size_t)b * xstride;
  const float* wr = W + (size_t)j * Hn;
  float s = bias[j];
  for (int k = 0; k < Hn; ++k) s += xr[k] * wr[k];
  out[idx] = s;
}

__global__ __launch_bounds__(256)
void gru_gate_k(const float* __restrict__ gi, const float* __restrict__ gh,
                float* __restrict__ h) {
  int idx = blockIdx.x * 256 + threadIdx.x;
  if (idx >= Bn * Hn) return;
  int b = idx / Hn, j = idx % Hn;
  const float* gib = gi + (size_t)b * 3 * Hn;
  const float* ghb = gh + (size_t)b * 3 * Hn;
  float r = 1.f / (1.f + expf(-(gib[j] + ghb[j])));
  float z = 1.f / (1.f + expf(-(gib[Hn + j] + ghb[Hn + j])));
  float n = tanhf(gib[2 * Hn + j] + r * ghb[2 * Hn + j]);
  h[idx] = (1.f - z) * n + z * h[idx];
}

__global__ __launch_bounds__(256)
void gru_logits_k(const float* __restrict__ h, const float* __restrict__ Wf,
                  const float* __restrict__ bfc, float* __restrict__ out, int t) {
  int idx = blockIdx.x * 256 + threadIdx.x;
  if (idx >= Bn * Cn) return;
  int b = idx / Cn, c = idx % Cn;
  float s = bfc[c];
  for (int k = 0; k < Hn; ++k) s += h[(size_t)b * Hn + k] * Wf[(size_t)k * Cn + c];
  out[((size_t)b * Tn + t) * Cn + c] = s;
}

// ---------------- host launcher ----------------
static inline char* align256p(char* p) {
  return (char*)(((uintptr_t)p + 255) & ~(uintptr_t)255);
}

extern "C" void kernel_launch(void* const* d_in, const int* in_sizes, int n_in,
                              void* d_out, int out_size, void* d_ws, size_t ws_size,
                              hipStream_t stream) {
  (void)in_sizes; (void)n_in; (void)out_size; (void)ws_size;
  const int*   input_ids = (const int*)d_in[0];
  const int*   attn_mask = (const int*)d_in[1];
  const float* emb_word  = (const float*)d_in[2];
  const float* emb_pos   = (const float*)d_in[3];
  const float* emb_tok   = (const float*)d_in[4];
  const float* ln_emb_g  = (const float*)d_in[5];
  const float* ln_emb_b  = (const float*)d_in[6];
  const float* Wq = (const float*)d_in[7];
  const float* Wk = (const float*)d_in[8];
  const float* Wv = (const float*)d_in[9];
  const float* Wo = (const float*)d_in[10];
  const float* bq = (const float*)d_in[11];
  const float* bk = (const float*)d_in[12];
  const float* bv = (const float*)d_in[13];
  const float* bo = (const float*)d_in[14];
  const float* ln1_g = (const float*)d_in[15];
  const float* ln1_b = (const float*)d_in[16];
  const float* W1 = (const float*)d_in[17];
  const float* b1 = (const float*)d_in[18];
  const float* W2 = (const float*)d_in[19];
  const float* b2 = (const float*)d_in[20];
  const float* ln2_g = (const float*)d_in[21];
  const float* ln2_b = (const float*)d_in[22];
  const float* W_ih = (const float*)d_in[23];
  const float* W_hh = (const float*)d_in[24];
  const float* b_ih = (const float*)d_in[25];
  const float* b_hh = (const float*)d_in[26];
  const float* Wf  = (const float*)d_in[27];
  const float* bfc = (const float*)d_in[28];
  float* out = (float*)d_out;

  // workspace carve (~97 MB)
  char* p = (char*)d_ws;
  auto take = [&](size_t bytes) -> char* { char* r = p; p = align256p(p + bytes); return r; };
  float* x    = (float*)take((size_t)NTOK * Hn * 4);
  bf16*  xb   = (bf16*) take((size_t)NTOK * Hn * 2);
  bf16*  qb   = (bf16*) take((size_t)NTOK * Hn * 2);
  bf16*  kb   = (bf16*) take((size_t)NTOK * Hn * 2);
  bf16*  vb   = (bf16*) take((size_t)NTOK * Hn * 2);
  bf16*  ctxb = (bf16*) take((size_t)NTOK * Hn * 2);
  float* tmp  = (float*)take((size_t)NTOK * Hn * 4);
  bf16*  hb   = (bf16*) take((size_t)NTOK * FFn * 2);
  bf16*  Wqb  = (bf16*) take((size_t)Hn * Hn * 2);
  bf16*  Wkb  = (bf16*) take((size_t)Hn * Hn * 2);
  bf16*  Wvb  = (bf16*) take((size_t)Hn * Hn * 2);
  bf16*  Wob  = (bf16*) take((size_t)Hn * Hn * 2);
  bf16*  W1b  = (bf16*) take((size_t)Hn * FFn * 2);
  bf16*  W2b  = (bf16*) take((size_t)FFn * Hn * 2);
  float* dec  = (float*)take((size_t)Hn * 4);
  float* hgru = (float*)take((size_t)Bn * Hn * 4);
  float* gi   = (float*)take((size_t)Bn * 3 * Hn * 4);
  float* gh   = (float*)take((size_t)Bn * 3 * Hn * 4);

  // embeddings + LN
  embed_ln_k<<<NTOK, 256, 0, stream>>>(input_ids, emb_word, emb_pos, emb_tok,
                                       ln_emb_g, ln_emb_b, x, xb);

  const int nHH = Hn * Hn, nHF = Hn * FFn;
  dim3 gHH(Hn / 128, NTOK / 128);    // (6, 32)
  dim3 gHF(FFn / 128, NTOK / 128);   // (24, 32)

  for (int l = 0; l < Ln; ++l) {
    const size_t oHH = (size_t)l * nHH, oHF = (size_t)l * nHF;
    cvt_bf16_k<<<(nHH + 255) / 256, 256, 0, stream>>>(Wq + oHH, Wqb, nHH);
    cvt_bf16_k<<<(nHH + 255) / 256, 256, 0, stream>>>(Wk + oHH, Wkb, nHH);
    cvt_bf16_k<<<(nHH + 255) / 256, 256, 0, stream>>>(Wv + oHH, Wvb, nHH);
    cvt_bf16_k<<<(nHH + 255) / 256, 256, 0, stream>>>(Wo + oHH, Wob, nHH);
    cvt_bf16_k<<<(nHF + 255) / 256, 256, 0, stream>>>(W1 + oHF, W1b, nHF);
    cvt_bf16_k<<<(nHF + 255) / 256, 256, 0, stream>>>(W2 + oHF, W2b, nHF);

    gemm_bf16_k<<<gHH, 256, 0, stream>>>(xb, Wqb, bq + (size_t)l * Hn, nullptr, qb,
                                         NTOK, Hn, Hn, 0);
    gemm_bf16_k<<<gHH, 256, 0, stream>>>(xb, Wkb, bk + (size_t)l * Hn, nullptr, kb,
                                         NTOK, Hn, Hn, 0);
    gemm_bf16_k<<<gHH, 256, 0, stream>>>(xb, Wvb, bv + (size_t)l * Hn, nullptr, vb,
                                         NTOK, Hn, Hn, 0);
    attention_k<<<Bn * NHn, 128, ATT_LDS, stream>>>(qb, kb, vb, attn_mask, ctxb);
    gemm_bf16_k<<<gHH, 256, 0, stream>>>(ctxb, Wob, bo + (size_t)l * Hn, tmp, nullptr,
                                         NTOK, Hn, Hn, 0);
    add_ln_k<<<NTOK, 256, 0, stream>>>(x, tmp, ln1_g + (size_t)l * Hn,
                                       ln1_b + (size_t)l * Hn, x, xb);
    gemm_bf16_k<<<gHF, 256, 0, stream>>>(xb, W1b, b1 + (size_t)l * FFn, nullptr, hb,
                                         NTOK, FFn, Hn, 1);
    gemm_bf16_k<<<gHH, 256, 0, stream>>>(hb, W2b, b2 + (size_t)l * Hn, tmp, nullptr,
                                         NTOK, Hn, FFn, 0);
    add_ln_k<<<NTOK, 256, 0, stream>>>(x, tmp, ln2_g + (size_t)l * Hn,
                                       ln2_b + (size_t)l * Hn, x, xb);
  }

  // GRU decoder
  decin_ln_k<<<1, 256, 0, stream>>>(emb_word, emb_pos, emb_tok, ln_emb_g, ln_emb_b, dec);
  gather_hidden_k<<<(Bn * Hn + 255) / 256, 256, 0, stream>>>(x, hgru);
  for (int t = 0; t < Tn; ++t) {
    const float* xt = (t == 0) ? dec : hgru;
    int xs = (t == 0) ? 0 : Hn;
    gru_gemm_k<<<(Bn * 3 * Hn + 255) / 256, 256, 0, stream>>>(xt, xs, W_ih, b_ih, gi);
    gru_gemm_k<<<(Bn * 3 * Hn + 255) / 256, 256, 0, stream>>>(hgru, Hn, W_hh, b_hh, gh);
    gru_gate_k<<<(Bn * Hn + 255) / 256, 256, 0, stream>>>(gi, gh, hgru);
    gru_logits_k<<<(Bn * Cn + 255) / 256, 256, 0, stream>>>(hgru, Wf, bfc, out, t);
  }
}
